// LSTM_12979391168907
// MI455X (gfx1250) — matrix operand order repaired and checked by reading, not compiled, not run on hardware
//
#include <hip/hip_runtime.h>

typedef __attribute__((ext_vector_type(16))) _Float16 v16h;
typedef __attribute__((ext_vector_type(8)))  _Float16 v8h;
typedef __attribute__((ext_vector_type(8)))  float    v8f;
typedef __attribute__((ext_vector_type(4)))  float    v4f;

#define HID   128
#define NIN   13
#define BATCH 262144
#define FOURH 512

#define LH_STRIDE 136   // halves per LDS row for h tile (16B-aligned sub-rows, de-conflicted banks)
#define LX_STRIDE 40    // halves per LDS row for x tile (K padded 13 -> 32 with zeros)

// ---------------------------------------------------------------------------
// Prep: convert weights to f16 WMMA B-fragments in workspace, fuse biases.
// B-fragment layout this code uses (16x16x32 f16, matched by A): lane n -> column (n&15);
// lanes 0-15 hold K=0..15 in halves 0..15, lanes 16-31 hold K=16..31.
// Fragment storage: [n_tile*4 + k_chunk] blocks of 512 halves, lane-major.
// ---------------------------------------------------------------------------
__global__ void lstm_prep(const float* __restrict__ Wxi, const float* __restrict__ Whi,
                          const float* __restrict__ Wxf, const float* __restrict__ Whf_,
                          const float* __restrict__ Wxg, const float* __restrict__ Whg,
                          const float* __restrict__ Wxo, const float* __restrict__ Who,
                          const float* __restrict__ bxi, const float* __restrict__ bhi,
                          const float* __restrict__ bxf, const float* __restrict__ bhf,
                          const float* __restrict__ bxg, const float* __restrict__ bhg,
                          const float* __restrict__ bxo, const float* __restrict__ bho,
                          _Float16* __restrict__ whf, _Float16* __restrict__ wxf,
                          float* __restrict__ bias)
{
    const float* Wh[4] = {Whi, Whf_, Whg, Who};
    const float* Wx[4] = {Wxi, Wxf, Wxg, Wxo};
    const float* bx[4] = {bxi, bxf, bxg, bxo};
    const float* bh[4] = {bhi, bhf, bhg, bho};
    const int tid = blockIdx.x * blockDim.x + threadIdx.x;
    const int nthr = gridDim.x * blockDim.x;

    // Wh fragments: 32 n_tiles x 4 k_chunks x (32 lanes x 16 halves)
    for (int e = tid; e < 32 * 4 * 512; e += nthr) {
        int frag = e >> 9, pos = e & 511;
        int lane = pos >> 4, hh = pos & 15;
        int nt = frag >> 2, kc = frag & 3;
        int K = hh + ((lane & 16) ? 16 : 0);
        int k = kc * 32 + K;
        int j = nt * 16 + (lane & 15);     // column in 4H pre-activation
        int gate = j >> 7, col = j & 127;
        whf[e] = (_Float16)Wh[gate][col * HID + k];
    }
    // Wx fragments: 32 n_tiles x (32 lanes x 16 halves), K padded to 32 with zeros
    for (int e = tid; e < 32 * 512; e += nthr) {
        int nt = e >> 9, pos = e & 511;
        int lane = pos >> 4, hh = pos & 15;
        int K = hh + ((lane & 16) ? 16 : 0);
        int j = nt * 16 + (lane & 15);
        int gate = j >> 7, col = j & 127;
        wxf[e] = (K < NIN) ? (_Float16)Wx[gate][col * NIN + K] : (_Float16)0.0f;
    }
    // fused bias
    for (int j = tid; j < FOURH; j += nthr) {
        int gate = j >> 7, col = j & 127;
        bias[j] = bx[gate][col] + bh[gate][col];
    }
}

// ---------------------------------------------------------------------------
// Main: one block = 16 batch rows. 8 waves, wave cg owns columns cg*16..+15 of
// every gate (n_tiles gate*8+cg). 5 WMMAs per gate (4 chunks of H + 1 of X).
// ---------------------------------------------------------------------------
__global__ void __launch_bounds__(256, 2) lstm_main(
    const float* __restrict__ x_in, const float* __restrict__ h_in,
    const float* __restrict__ c_in,
    const _Float16* __restrict__ whf, const _Float16* __restrict__ wxf,
    const float* __restrict__ bias,
    float* __restrict__ out)
{
    __shared__ __align__(16) _Float16 lh[16 * LH_STRIDE];
    __shared__ __align__(16) _Float16 lx[16 * LX_STRIDE];

    const int t = threadIdx.x;
    const long m0 = (long)blockIdx.x * 16;

    // Stage h tile: 16 rows x 128 f32 -> f16 LDS, coalesced float4 loads
    {
        const v4f* src = (const v4f*)(h_in + m0 * HID);
#pragma unroll
        for (int i = 0; i < 2; ++i) {
            int idx = t + i * 256;              // 0..511 float4s
            int row = idx >> 5, c4 = idx & 31;  // 32 float4 per row
            v4f v = src[row * 32 + c4];
            _Float16* dst = &lh[row * LH_STRIDE + c4 * 4];
            dst[0] = (_Float16)v[0]; dst[1] = (_Float16)v[1];
            dst[2] = (_Float16)v[2]; dst[3] = (_Float16)v[3];
        }
    }
    // Stage x tile: 16 rows x 13 f32 -> f16 LDS padded to 32 K with zeros
    for (int idx = t; idx < 16 * LX_STRIDE; idx += 256) {
        int row = idx / LX_STRIDE, col = idx - row * LX_STRIDE;
        float v = (col < NIN) ? x_in[(m0 + row) * NIN + col] : 0.0f;
        lx[idx] = (_Float16)v;
    }
    __syncthreads();

    const int lane = t & 31;
    const int cg   = t >> 5;          // 0..7: column group within each gate
    const int n    = lane & 15;
    const int hi16 = (lane >> 4) & 1;

    // A fragments (16x32 f16 per chunk), K order matching B fragments:
    //   lanes 0-15 : halves 0..15 -> K=0..15
    //   lanes 16-31: halves 0..15 -> K=16..31
    v16h a[5];
    {
        const int arow = n * LH_STRIDE;
#pragma unroll
        for (int kc = 0; kc < 4; ++kc) {
            v8h lo = *(const v8h*)&lh[arow + kc * 32 + hi16 * 16];
            v8h hi = *(const v8h*)&lh[arow + kc * 32 + hi16 * 16 + 8];
#pragma unroll
            for (int q = 0; q < 8; ++q) { a[kc][q] = lo[q]; a[kc][8 + q] = hi[q]; }
        }
        const int xrow = n * LX_STRIDE;
        v8h lo = *(const v8h*)&lx[xrow + hi16 * 16];
        v8h hi = *(const v8h*)&lx[xrow + hi16 * 16 + 8];
#pragma unroll
        for (int q = 0; q < 8; ++q) { a[4][q] = lo[q]; a[4][8 + q] = hi[q]; }
    }

    v8f acc[4];
#pragma unroll
    for (int g = 0; g < 4; ++g) {
        const int nt = g * 8 + cg;
        v8f c = {};
#pragma unroll
        for (int kc = 0; kc < 4; ++kc) {
            v16h b = *(const v16h*)(whf + ((nt * 4 + kc) << 9) + lane * 16);
            c = __builtin_amdgcn_wmma_f32_16x16x32_f16(
                    false, a[kc], false, b, (short)0, c, false, false);
        }
        v16h bxm = *(const v16h*)(wxf + (nt << 9) + lane * 16);
        c = __builtin_amdgcn_wmma_f32_16x16x32_f16(
                false, a[4], false, bxm, (short)0, c, false, false);
        acc[g] = c;
    }

    float bvals[4];
#pragma unroll
    for (int g = 0; g < 4; ++g) bvals[g] = bias[g * HID + cg * 16 + n];

    // D layout: VGPR v, lanes 0-15 -> row v, lanes 16-31 -> row v+8; col = lane&15
    const long outHC = (long)BATCH * HID;
#pragma unroll
    for (int v = 0; v < 8; ++v) {
        long b   = m0 + v + hi16 * 8;
        long off = b * HID + cg * 16 + n;
        float pi = acc[0][v] + bvals[0];
        float pf = acc[1][v] + bvals[1];
        float pg = acc[2][v] + bvals[2];
        float po = acc[3][v] + bvals[3];
        float it = 1.0f / (1.0f + __expf(-pi));
        float ft = 1.0f / (1.0f + __expf(-pf));
        float gt = fminf(fmaxf(pg, -1.0f), 1.0f);
        float ot = 1.0f / (1.0f + __expf(-po));
        float ct = ft * c_in[off] + it * gt;
        float ht = ot * fminf(fmaxf(ct, -1.0f), 1.0f);
        out[off]             = ht;   // h_t
        out[outHC + off]     = ht;   // h_t (second tuple element)
        out[2 * outHC + off] = ct;   // c_t
    }
}

extern "C" void kernel_launch(void* const* d_in, const int* in_sizes, int n_in,
                              void* d_out, int out_size, void* d_ws, size_t ws_size,
                              hipStream_t stream)
{
    const float* x_in = (const float*)d_in[0];
    const float* h_in = (const float*)d_in[1];
    const float* c_in = (const float*)d_in[2];
    const float* W_xi = (const float*)d_in[3];
    const float* b_xi = (const float*)d_in[4];
    const float* W_hi = (const float*)d_in[5];
    const float* b_hi = (const float*)d_in[6];
    const float* W_xf = (const float*)d_in[7];
    const float* b_xf = (const float*)d_in[8];
    const float* W_hf = (const float*)d_in[9];
    const float* b_hf = (const float*)d_in[10];
    const float* W_xg = (const float*)d_in[11];
    const float* b_xg = (const float*)d_in[12];
    const float* W_hg = (const float*)d_in[13];
    const float* b_hg = (const float*)d_in[14];
    const float* W_xo = (const float*)d_in[15];
    const float* b_xo = (const float*)d_in[16];
    const float* W_ho = (const float*)d_in[17];
    const float* b_ho = (const float*)d_in[18];

    _Float16* whf  = (_Float16*)d_ws;            // 32*4*512 halves = 128 KB
    _Float16* wxf  = whf + 32 * 4 * 512;         // 32*512 halves   =  32 KB
    float*    bias = (float*)(wxf + 32 * 512);   // 512 floats      =   2 KB

    lstm_prep<<<64, 256, 0, stream>>>(W_xi, W_hi, W_xf, W_hf, W_xg, W_hg, W_xo, W_ho,
                                      b_xi, b_hi, b_xf, b_hf, b_xg, b_hg, b_xo, b_ho,
                                      whf, wxf, bias);
    lstm_main<<<BATCH / 16, 256, 0, stream>>>(x_in, h_in, c_in, whf, wxf, bias,
                                              (float*)d_out);
}
